// MainNet_55662776156502
// MI455X (gfx1250) — matrix operand order repaired and checked by reading, not compiled, not run on hardware
//
#include <hip/hip_runtime.h>
#include <hip/hip_bf16.h>

typedef __attribute__((ext_vector_type(16))) _Float16 v16h;
typedef __attribute__((ext_vector_type(8)))  _Float16 v8h;
typedef __attribute__((ext_vector_type(8)))  float    v8f;

// Problem constants: b=2, an_sparse=2, an_novel=7, rp=32, ps=63, K=32, N=1024
// groups = b*an_novel*an_sparse*N = 28672

// ---------------------------------------------------------------------------
// Pack MLP weights into WMMA B-fragment layout:
//   fragment (ks, nt): lane holds column c = nt*16 + lane%16,
//   v16h element j holds K = ks*32 + (lane/16)*8 + (j%8) + (j/8)*16.
// ---------------------------------------------------------------------------
__global__ __launch_bounds__(256) void pack_weights_kernel(
    const float* __restrict__ w1,   // (67,64)
    const float* __restrict__ w2,   // (64,64)
    _Float16* __restrict__ p1,      // 3*4*32*16 halves
    _Float16* __restrict__ p2)      // 2*4*32*16 halves
{
  int t = blockIdx.x * blockDim.x + threadIdx.x;
  const int total1 = 3 * 4 * 32 * 16;
  const int total2 = 2 * 4 * 32 * 16;
  if (t < total1) {
    int j = t & 15, lane = (t >> 4) & 31, nt = (t >> 9) & 3, ks = t >> 11;
    int K = ks * 32 + ((lane >> 4) << 3) + (j & 7) + ((j >> 3) << 4);
    int c = nt * 16 + (lane & 15);
    float v = (K < 67) ? w1[K * 64 + c] : 0.0f;
    p1[t] = (_Float16)v;
  }
  if (t < total2) {
    int j = t & 15, lane = (t >> 4) & 31, nt = (t >> 9) & 3, ks = t >> 11;
    int K = ks * 32 + ((lane >> 4) << 3) + (j & 7) + ((j >> 3) << 4);
    int c = nt * 16 + (lane & 15);
    p2[t] = (_Float16)w2[K * 64 + c];
  }
}

// ---------------------------------------------------------------------------
// Fill static columns of the augmented feature tensor augG[(m,h,x)][72]:
//   col 0 = flow value, cols 65..71 = 0 (cols 1..64 are written by conv3).
// ---------------------------------------------------------------------------
__global__ __launch_bounds__(256) void aug_fill_kernel(
    const float* __restrict__ flow,  // (4,32,63) flat
    _Float16* __restrict__ augG)     // (4*32*63, 72) halves
{
  int i = blockIdx.x * blockDim.x + threadIdx.x;
  if (i >= 4 * 32 * 63) return;
  _Float16* row = augG + (size_t)i * 72;
  row[0] = (_Float16)flow[i];
  #pragma unroll
  for (int c = 65; c < 72; ++c) row[c] = (_Float16)0.0f;
}

// ---------------------------------------------------------------------------
// 3x3 SAME conv, 3 input planes -> 64ch, ReLU (tiny: 27 MACs/output, direct)
// ---------------------------------------------------------------------------
__global__ __launch_bounds__(256) void conv3_in3_relu(
    const float* __restrict__ p0, const float* __restrict__ p1,
    const float* __restrict__ p2,
    const float* __restrict__ w,     // (64,3,3,3)
    const float* __restrict__ bias,  // (64)
    float* __restrict__ out, int M, int H, int W)
{
  int idx = blockIdx.x * blockDim.x + threadIdx.x;
  int total = M * 64 * H * W;
  if (idx >= total) return;
  int x  = idx % W;
  int y  = (idx / W) % H;
  int co = (idx / (W * H)) % 64;
  int m  = idx / (W * H * 64);
  const float* planes[3] = { p0 + (size_t)m * H * W,
                             p1 + (size_t)m * H * W,
                             p2 + (size_t)m * H * W };
  float acc = bias[co];
  #pragma unroll
  for (int ci = 0; ci < 3; ++ci) {
    const float* ip = planes[ci];
    const float* wp = w + (co * 3 + ci) * 9;
    #pragma unroll
    for (int dy = -1; dy <= 1; ++dy) {
      int yy = y + dy;
      if (yy < 0 || yy >= H) continue;
      #pragma unroll
      for (int dx = -1; dx <= 1; ++dx) {
        int xx = x + dx;
        if (xx < 0 || xx >= W) continue;
        acc += ip[yy * W + xx] * wp[(dy + 1) * 3 + (dx + 1)];
      }
    }
  }
  out[idx] = fmaxf(acc, 0.0f);
}

// ---------------------------------------------------------------------------
// 3x3 SAME conv 64 -> 64, ReLU. One block per (m, y) output row; the
// 64ci x 3row x (W+2) zero-padded input slab staged once into LDS (f16).
// ---------------------------------------------------------------------------
__global__ __launch_bounds__(256) void conv3_64_relu(
    const float* __restrict__ in,    // (M,64,H,W)
    const float* __restrict__ w,     // (64,64,3,3)
    const float* __restrict__ bias,  // (64)
    float* __restrict__ out, int M, int H, int W)
{
  __shared__ _Float16 slab[64 * 3 * 65];   // max W=63 -> WP=65 (25 KB)
  int y = blockIdx.x % H;
  int m = blockIdx.x / H;
  int WP = W + 2;
  int tid = threadIdx.x;

  for (int i = tid; i < 64 * 3 * WP; i += 256) {
    int ci  = i / (3 * WP);
    int rem = i - ci * 3 * WP;
    int ry  = rem / WP;
    int xx  = rem - ry * WP;
    int yy  = y + ry - 1;
    int xs  = xx - 1;
    float vv = 0.0f;
    if (yy >= 0 && yy < H && xs >= 0 && xs < W)
      vv = in[(((size_t)m * 64 + ci) * H + yy) * W + xs];
    slab[i] = (_Float16)vv;
  }
  __syncthreads();

  for (int i = tid; i < 64 * W; i += 256) {
    int co = i / W;
    int x  = i - co * W;
    float acc = bias[co];
    const float* wp = w + (size_t)co * 64 * 9;
    for (int ci = 0; ci < 64; ++ci) {
      const _Float16* sp = &slab[(ci * 3) * WP + x];
      const float* wc = wp + ci * 9;
      #pragma unroll
      for (int ry = 0; ry < 3; ++ry) {
        #pragma unroll
        for (int dx = 0; dx < 3; ++dx)
          acc += (float)sp[ry * WP + dx] * wc[ry * 3 + dx];
      }
    }
    out[(((size_t)m * 64 + co) * H + y) * W + x] = fmaxf(acc, 0.0f);
  }
}

// ---------------------------------------------------------------------------
// Same as conv3_64_relu, but writes output as f16 into the augmented layout
// augG[((m*32+y)*63+x)*72 + co + 1]  (feature-net conv3 only; M=4,H=32,W=63).
// ---------------------------------------------------------------------------
__global__ __launch_bounds__(256) void conv3_64_to_aug(
    const float* __restrict__ in,    // (M,64,H,W)
    const float* __restrict__ w,     // (64,64,3,3)
    const float* __restrict__ bias,  // (64)
    _Float16* __restrict__ augG, int M, int H, int W)
{
  __shared__ _Float16 slab[64 * 3 * 65];
  int y = blockIdx.x % H;
  int m = blockIdx.x / H;
  int WP = W + 2;
  int tid = threadIdx.x;

  for (int i = tid; i < 64 * 3 * WP; i += 256) {
    int ci  = i / (3 * WP);
    int rem = i - ci * 3 * WP;
    int ry  = rem / WP;
    int xx  = rem - ry * WP;
    int yy  = y + ry - 1;
    int xs  = xx - 1;
    float vv = 0.0f;
    if (yy >= 0 && yy < H && xs >= 0 && xs < W)
      vv = in[(((size_t)m * 64 + ci) * H + yy) * W + xs];
    slab[i] = (_Float16)vv;
  }
  __syncthreads();

  for (int i = tid; i < 64 * W; i += 256) {
    int co = i / W;
    int x  = i - co * W;
    float acc = bias[co];
    const float* wp = w + (size_t)co * 64 * 9;
    for (int ci = 0; ci < 64; ++ci) {
      const _Float16* sp = &slab[(ci * 3) * WP + x];
      const float* wc = wp + ci * 9;
      #pragma unroll
      for (int ry = 0; ry < 3; ++ry) {
        #pragma unroll
        for (int dx = 0; dx < 3; ++dx)
          acc += (float)sp[ry * WP + dx] * wc[ry * 3 + dx];
      }
    }
    augG[((size_t)(m * H + y) * W + x) * 72 + co + 1] =
        (_Float16)fmaxf(acc, 0.0f);
  }
}

// ---------------------------------------------------------------------------
// 3x3 SAME conv 64 -> 1, no ReLU, plus residual add of pn (tiny)
// ---------------------------------------------------------------------------
__global__ __launch_bounds__(256) void conv3_64to1_res(
    const float* __restrict__ in,    // (M,64,H,W)
    const float* __restrict__ w,     // (1,64,3,3)
    const float* __restrict__ bias,  // (1)
    const float* __restrict__ pn,    // (M,H,W)
    float* __restrict__ out, int M, int H, int W)
{
  int idx = blockIdx.x * blockDim.x + threadIdx.x;
  int total = M * H * W;
  if (idx >= total) return;
  int x = idx % W;
  int y = (idx / W) % H;
  int m = idx / (W * H);
  const float* ib = in + (size_t)m * 64 * H * W;
  float acc = bias[0];
  for (int ci = 0; ci < 64; ++ci) {
    const float* ip = ib + (size_t)ci * H * W;
    const float* wp = w + ci * 9;
    #pragma unroll
    for (int dy = -1; dy <= 1; ++dy) {
      int yy = y + dy;
      if (yy < 0 || yy >= H) continue;
      #pragma unroll
      for (int dx = -1; dx <= 1; ++dx) {
        int xx = x + dx;
        if (xx < 0 || xx >= W) continue;
        acc += ip[yy * W + xx] * wp[(dy + 1) * 3 + (dx + 1)];
      }
    }
  }
  out[idx] = pn[idx] + acc;
}

// ---------------------------------------------------------------------------
// Fused MLP + softmax(K) + EPI dot.
// Block per (m,h): the pre-converted augmented strip (9072 B, f16) is copied
// global->LDS with async-to-LDS b128 ops (ASYNCcnt path), then 7 waves
// (one per novel view) sweep ow=0..31; layers 1&2 use
// v_wmma_f32_16x16x32_f16 with ds_read_b128 A-fragments and VGPR-resident
// packed weights.
// ---------------------------------------------------------------------------
__global__ __launch_bounds__(224) void mlp_epi_kernel(
    const _Float16* __restrict__ augG, // (4*32*63, 72) halves
    const float* __restrict__ lf,      // (4,32,63)
    const _Float16* __restrict__ pw1,  // packed W1 fragments
    const _Float16* __restrict__ pw2,  // packed W2 fragments
    const float* __restrict__ b1,      // (64)
    const float* __restrict__ b2,      // (64)
    const float* __restrict__ w3v,     // (64)
    const float* __restrict__ b3p,     // (1)
    const float* __restrict__ cw,      // (64)
    const float* __restrict__ cbp,     // (1)
    float* __restrict__ nv_out,        // (28672)
    float* __restrict__ conf_out)      // (28672)
{
  __shared__ __attribute__((aligned(16))) _Float16 aug[63 * 72];      //  9072 B
  __shared__ __attribute__((aligned(16))) _Float16 h1s[7 * 16 * 72];  // 16128 B
  __shared__ __attribute__((aligned(16))) _Float16 h2s[7 * 32 * 72];  // 32256 B
  __shared__ float w3s[64];
  __shared__ float cws[64];
  __shared__ float lf_s[63];

  const int tid = threadIdx.x;
  const int m  = blockIdx.x >> 5;   // 0..3   (b*an_sparse)
  const int h  = blockIdx.x & 31;   // 0..31
  const int bb = m >> 1;
  const int s  = m & 1;

  // ---- async global->LDS copy of the augmented strip (567 x 16B) ----
  {
    const _Float16* strip = augG + (size_t)(m * 32 + h) * 4536; // 63*72
    unsigned ldsBase = (unsigned)(uintptr_t)(&aug[0]);
    for (int i = tid; i < 567; i += 224) {
      unsigned off = (unsigned)i * 16u;
      unsigned ldsAddr = ldsBase + off;
      asm volatile("global_load_async_to_lds_b128 %0, %1, %2"
                   :: "v"(ldsAddr), "v"(off), "s"(strip) : "memory");
    }
  }

  const float* lfp = lf + (m * 32 + h) * 63;
  for (int i = tid; i < 63; i += 224) lf_s[i] = lfp[i];
  if (tid < 64) { w3s[tid] = w3v[tid]; cws[tid] = cw[tid]; }

  asm volatile("s_wait_asynccnt 0x0" ::: "memory");
  __syncthreads();

  const int wave = tid >> 5;        // 0..6 == novel view v
  const int lane = tid & 31;
  const int half = lane >> 4;
  const int ln   = lane & 15;
  const int v    = wave;
  const float angv = (s == 0) ? -(float)(v + 1) : (float)(7 - v);

  float bias1[4], bias2[4];
  #pragma unroll
  for (int nt = 0; nt < 4; ++nt) { bias1[nt] = b1[nt * 16 + ln]; bias2[nt] = b2[nt * 16 + ln]; }
  const float b3 = b3p[0];
  const float cb = cbp[0];

  _Float16* h1w = &h1s[wave * 16 * 72];
  _Float16* h2w = &h2s[wave * 32 * 72];

  const v16h* PW1 = reinterpret_cast<const v16h*>(pw1);
  const v16h* PW2 = reinterpret_cast<const v16h*>(pw2);

  const int gbase = ((bb * 7 + v) * 2 + s) * 1024 + h * 32;
  const _Float16 zh = (_Float16)0.0f;

  for (int ow = 0; ow < 32; ++ow) {
    for (int mtile = 0; mtile < 2; ++mtile) {
      const int kk = mtile * 16 + ln;        // candidate index (A row)
      const _Float16* arow = &aug[(ow + kk) * 72];

      // ---------------- layer 1: [16 x 96(pad67)] x [96 x 64] --------------
      v8f acc[4] = {};
      #pragma unroll
      for (int ks = 0; ks < 3; ++ks) {
        v16h a;
        if (ks < 2) {
          int c0 = ks * 32 + half * 8;
          v8h lo = *reinterpret_cast<const v8h*>(&arow[c0]);
          v8h hi = *reinterpret_cast<const v8h*>(&arow[c0 + 16]);
          #pragma unroll
          for (int j = 0; j < 8; ++j) { a[j] = lo[j]; a[j + 8] = hi[j]; }
        } else {
          v8h lo = *reinterpret_cast<const v8h*>(&arow[64]); // [feat63,0,..,0]
          #pragma unroll
          for (int j = 0; j < 8; ++j) a[j] = (half == 0) ? lo[j] : zh;
          a[1] = (half == 0) ? (_Float16)(float)(kk - 16) : zh;  // spatial
          a[2] = (half == 0) ? (_Float16)angv : zh;              // angular
          #pragma unroll
          for (int j = 8; j < 16; ++j) a[j] = zh;
        }
        #pragma unroll
        for (int nt = 0; nt < 4; ++nt) {
          v16h bf = PW1[(ks * 4 + nt) * 32 + lane];
          acc[nt] = __builtin_amdgcn_wmma_f32_16x16x32_f16(
              false, a, false, bf, (short)0, acc[nt], false, false);
        }
      }
      #pragma unroll
      for (int nt = 0; nt < 4; ++nt) {
        #pragma unroll
        for (int g = 0; g < 8; ++g) {
          int row = g + half * 8;
          float x = acc[nt][g] + bias1[nt];
          h1w[row * 72 + nt * 16 + ln] = (_Float16)fmaxf(x, 0.0f);
        }
      }
      asm volatile("s_wait_dscnt 0x0" ::: "memory");

      // ---------------- layer 2: [16 x 64] x [64 x 64] ----------------------
      v8f acc2[4] = {};
      #pragma unroll
      for (int ks = 0; ks < 2; ++ks) {
        int c0 = ks * 32 + half * 8;
        v8h lo = *reinterpret_cast<const v8h*>(&h1w[ln * 72 + c0]);
        v8h hi = *reinterpret_cast<const v8h*>(&h1w[ln * 72 + c0 + 16]);
        v16h a;
        #pragma unroll
        for (int j = 0; j < 8; ++j) { a[j] = lo[j]; a[j + 8] = hi[j]; }
        #pragma unroll
        for (int nt = 0; nt < 4; ++nt) {
          v16h bf = PW2[(ks * 4 + nt) * 32 + lane];
          acc2[nt] = __builtin_amdgcn_wmma_f32_16x16x32_f16(
              false, a, false, bf, (short)0, acc2[nt], false, false);
        }
      }
      #pragma unroll
      for (int nt = 0; nt < 4; ++nt) {
        #pragma unroll
        for (int g = 0; g < 8; ++g) {
          int row = mtile * 16 + g + half * 8;
          float x = acc2[nt][g] + bias2[nt];
          h2w[row * 72 + nt * 16 + ln] = (_Float16)fmaxf(x, 0.0f);
        }
      }
    }
    asm volatile("s_wait_dscnt 0x0" ::: "memory");

    // ------ layer 3 + conf + softmax over K + EPI dot (lane == k) -----------
    const int k = lane;
    float aw = 0.0f, ac = 0.0f;
    #pragma unroll
    for (int cc = 0; cc < 8; ++cc) {
      v8h hx = *reinterpret_cast<const v8h*>(&h2w[k * 72 + cc * 8]);
      #pragma unroll
      for (int j = 0; j < 8; ++j) {
        float x = (float)hx[j];
        aw += x * w3s[cc * 8 + j];
        ac += x * cws[cc * 8 + j];
      }
    }
    float wgt = aw + b3;
    float mx = wgt;
    #pragma unroll
    for (int off = 16; off >= 1; off >>= 1) mx = fmaxf(mx, __shfl_xor(mx, off));
    float e = __expf(wgt - mx);
    float se = e;
    #pragma unroll
    for (int off = 16; off >= 1; off >>= 1) se += __shfl_xor(se, off);
    float p = e / se;
    float nvv = p * lf_s[ow + k];
    float cs = ac;
    #pragma unroll
    for (int off = 16; off >= 1; off >>= 1) {
      nvv += __shfl_xor(nvv, off);
      cs  += __shfl_xor(cs, off);
    }
    if (lane == 0) {
      nv_out[gbase + ow]   = nvv;
      conf_out[gbase + ow] = cs * (1.0f / 32.0f) + cb;
    }
  }
}

// ---------------------------------------------------------------------------
// Confidence softmax over an_sparse=2 and blend: novel[b,v,n]
// ---------------------------------------------------------------------------
__global__ __launch_bounds__(256) void combine_kernel(
    const float* __restrict__ nv, const float* __restrict__ conf,
    float* __restrict__ novel)
{
  int idx = blockIdx.x * blockDim.x + threadIdx.x;
  if (idx >= 14336) return;               // b*an_novel*N = 2*7*1024
  int n  = idx & 1023;
  int bv = idx >> 10;
  int g0 = (bv * 2 + 0) * 1024 + n;
  int g1 = (bv * 2 + 1) * 1024 + n;
  float c0 = conf[g0], c1 = conf[g1];
  float mx = fmaxf(c0, c1);
  float e0 = __expf(c0 - mx), e1 = __expf(c1 - mx);
  float inv = 1.0f / (e0 + e1);
  novel[idx] = nv[g0] * (e0 * inv) + nv[g1] * (e1 * inv);
}

// ---------------------------------------------------------------------------
extern "C" void kernel_launch(void* const* d_in, const int* in_sizes, int n_in,
                              void* d_out, int out_size, void* d_ws, size_t ws_size,
                              hipStream_t stream) {
  (void)in_sizes; (void)n_in; (void)out_size; (void)ws_size;

  const float* lf_sparse  = (const float*)d_in[0];
  const float* flow_sp    = (const float*)d_in[1];
  const float* warped     = (const float*)d_in[2];
  const float* patch_l    = (const float*)d_in[3];
  const float* patch_r    = (const float*)d_in[4];
  const float* fr_w1      = (const float*)d_in[5];
  const float* fr_b1      = (const float*)d_in[6];
  const float* fr_w2      = (const float*)d_in[7];
  const float* fr_b2      = (const float*)d_in[8];
  const float* fr_w3      = (const float*)d_in[9];
  const float* fr_b3      = (const float*)d_in[10];
  const float* mlp_w1     = (const float*)d_in[11];
  const float* mlp_b1     = (const float*)d_in[12];
  const float* mlp_w2     = (const float*)d_in[13];
  const float* mlp_b2     = (const float*)d_in[14];
  const float* mlp_w3     = (const float*)d_in[15];
  const float* mlp_b3     = (const float*)d_in[16];
  const float* conf_w     = (const float*)d_in[17];
  const float* conf_b     = (const float*)d_in[18];
  const float* vr_w1      = (const float*)d_in[19];
  const float* vr_b1      = (const float*)d_in[20];
  const float* vr_w2      = (const float*)d_in[21];
  const float* vr_b2      = (const float*)d_in[22];
  const float* vr_w3      = (const float*)d_in[23];
  const float* vr_b3      = (const float*)d_in[24];

  char* ws = (char*)d_ws;
  _Float16* augG  = (_Float16*)(ws + 0);              // 580608 halves (1.16 MB)
  float*    tmpA  = (float*)(ws + 2064384);           // up to 917504 f32
  float*    tmpB  = (float*)(ws + 5734400);           // up to 917504 f32
  float*    nvb   = (float*)(ws + 9404416);           // 28672 f32
  float*    confb = (float*)(ws + 9519104);           // 28672 f32
  float*    novel = (float*)(ws + 9633792);           // 14336 f32
  _Float16* pw1   = (_Float16*)(ws + 9691136);        // 6144 halves
  _Float16* pw2   = (_Float16*)(ws + 9703424);        // 4096 halves

  // 0) pack MLP weights; fill static columns of augmented tensor
  pack_weights_kernel<<<24, 256, 0, stream>>>(mlp_w1, mlp_w2, pw1, pw2);
  aug_fill_kernel<<<32, 256, 0, stream>>>(flow_sp, augG);

  // 1) feature-extraction conv net on (4,3,32,63); channels [flow, lf, warped]
  conv3_in3_relu<<<2016, 256, 0, stream>>>(flow_sp, lf_sparse, warped,
                                           fr_w1, fr_b1, tmpA, 4, 32, 63);
  conv3_64_relu<<<128, 256, 0, stream>>>(tmpA, fr_w2, fr_b2, tmpB, 4, 32, 63);
  conv3_64_to_aug<<<128, 256, 0, stream>>>(tmpB, fr_w3, fr_b3, augG, 4, 32, 63);

  // 2) fused WMMA MLP + softmax(K) + EPI dot: block per (m,h), wave per view
  mlp_epi_kernel<<<128, 224, 0, stream>>>(augG, lf_sparse, pw1, pw2,
                                          mlp_b1, mlp_b2, mlp_w3, mlp_b3,
                                          conf_w, conf_b, nvb, confb);

  // 3) confidence softmax over the 2 sparse views + blend
  combine_kernel<<<56, 256, 0, stream>>>(nvb, confb, novel);

  // 4) view-refinement conv net on (14,3,32,32); channels [novel, left, right]
  conv3_in3_relu<<<3584, 256, 0, stream>>>(novel, patch_l, patch_r,
                                           vr_w1, vr_b1, tmpA, 14, 32, 32);
  conv3_64_relu<<<448, 256, 0, stream>>>(tmpA, vr_w2, vr_b2, tmpB, 14, 32, 32);
  conv3_64to1_res<<<56, 256, 0, stream>>>(tmpB, vr_w3, vr_b3, novel,
                                          (float*)d_out, 14, 32, 32);
}
